// Model_46986942218876
// MI455X (gfx1250) — matrix operand order repaired and checked by reading, not compiled, not run on hardware
//
#include <hip/hip_runtime.h>
#include <hip/hip_bf16.h>

typedef __bf16 v16bf __attribute__((ext_vector_type(16)));
typedef __bf16 v8bf  __attribute__((ext_vector_type(8)));
typedef float  v8f   __attribute__((ext_vector_type(8)));

#define BATCH 16
#define SEQ   4096
#define DIM   64
#define BM    64   // query rows per block
#define BN    64   // keys per tile
#define NW    4    // waves per block (wave32)

__device__ __forceinline__ unsigned short f2bf(float x) {
    unsigned u = __builtin_bit_cast(unsigned, x);
    u += 0x7FFFu + ((u >> 16) & 1u);   // round-to-nearest-even
    return (unsigned short)(u >> 16);
}

__device__ __forceinline__ v16bf cat16(uint4 a, uint4 b) {
    v8bf lo = __builtin_bit_cast(v8bf, a);
    v8bf hi = __builtin_bit_cast(v8bf, b);
    return __builtin_shufflevector(lo, hi, 0,1,2,3,4,5,6,7,8,9,10,11,12,13,14,15);
}

// 32B contiguous (global or LDS) -> one bf16 fragment
__device__ __forceinline__ v16bf frag32B(const unsigned short* p) {
    uint4 a, b;
    __builtin_memcpy(&a, p, 16);
    __builtin_memcpy(&b, p + 8, 16);
    return cat16(a, b);
}
__device__ __forceinline__ v16bf frag2(const unsigned short* p0, const unsigned short* p1) {
    uint4 a, b;
    __builtin_memcpy(&a, p0, 16);
    __builtin_memcpy(&b, p1, 16);
    return cat16(a, b);
}

union BF16x16 { v16bf v; unsigned short u[16]; uint4 q[2]; };

// ============================ pre-pass kernels ============================
// K: f32 [b][s][d] -> bf16 [b][s][d]
__global__ __launch_bounds__(256)
void conv_k_bf16(const float* __restrict__ k, unsigned short* __restrict__ kbf) {
    long i = ((long)blockIdx.x * 256 + threadIdx.x) * 8;
    const float4* p = (const float4*)(k + i);
    float4 a = p[0], b = p[1];
    float f[8] = {a.x,a.y,a.z,a.w,b.x,b.y,b.z,b.w};
    unsigned short u[8];
    #pragma unroll
    for (int j = 0; j < 8; ++j) u[j] = f2bf(f[j]);
    __builtin_memcpy(kbf + i, u, 16);
}

// V: f32 [b][s][d] -> bf16 transposed [b][d][s]
__global__ __launch_bounds__(256)
void conv_v_t_bf16(const float* __restrict__ v, unsigned short* __restrict__ vtb) {
    __shared__ __align__(16) unsigned short sT[DIM * 64]; // [d][s-in-tile]
    const int b  = blockIdx.y;
    const int s0 = blockIdx.x * 64;
    const int t  = threadIdx.x;
    {   // read 64x64 f32 tile coalesced, scatter-transpose into LDS as bf16
        const int s = t >> 2, dp = (t & 3) * 16;
        const float4* p = (const float4*)(v + ((long)b * SEQ + s0 + s) * DIM + dp);
        #pragma unroll
        for (int h = 0; h < 4; ++h) {
            float4 x = p[h];
            sT[(dp + 4*h + 0) * 64 + s] = f2bf(x.x);
            sT[(dp + 4*h + 1) * 64 + s] = f2bf(x.y);
            sT[(dp + 4*h + 2) * 64 + s] = f2bf(x.z);
            sT[(dp + 4*h + 3) * 64 + s] = f2bf(x.w);
        }
    }
    __syncthreads();
    {   // write rows of V^T coalesced
        const int d = t >> 2, sp = (t & 3) * 16;
        uint4 w[2];
        __builtin_memcpy(w, &sT[d * 64 + sp], 32);
        __builtin_memcpy(vtb + ((long)b * DIM + d) * SEQ + s0 + sp, w, 32);
    }
}

// ===================== fast path: bf16 K/V^T in global =====================
__global__ __launch_bounds__(128)
void fa_fwd_fast(const float* __restrict__ q, const unsigned short* __restrict__ kbf,
                 const unsigned short* __restrict__ vtb, const float* __restrict__ inv_scale,
                 float* __restrict__ out)
{
    __shared__ __align__(16) unsigned short sP[NW * 16 * BN]; // per-wave P staging only

    const int b     = blockIdx.y;
    const int qbase = blockIdx.x * BM;
    const int tid   = threadIdx.x;
    const int wave  = tid >> 5;
    const int lane  = tid & 31;
    const int lrow  = lane & 15;
    const int lhalf = lane >> 4;

    const float qscale = (1.0f / inv_scale[0]) * 1.44269504088896340736f;

    // Q tile: two pre-scaled A-fragments, loaded once
    const float* qp = q + ((long)b * SEQ + qbase + wave * 16 + lrow) * DIM;
    v16bf qfrag[2];
    for (int c = 0; c < 2; ++c) {
        const float4* p0 = (const float4*)(qp + c * 32 + lhalf * 16);
        const float4* p1 = (const float4*)(qp + c * 32 + lhalf * 16 + 8);
        float4 a0 = p0[0], a1 = p0[1], b0 = p1[0], b1 = p1[1];
        float f0[8] = {a0.x,a0.y,a0.z,a0.w,a1.x,a1.y,a1.z,a1.w};
        float f1[8] = {b0.x,b0.y,b0.z,b0.w,b1.x,b1.y,b1.z,b1.w};
        BF16x16 t;
        #pragma unroll
        for (int i = 0; i < 8; ++i) {
            t.u[i]     = f2bf(f0[i] * qscale);
            t.u[8 + i] = f2bf(f1[i] * qscale);
        }
        qfrag[c] = t.v;
    }

    v16bf onesfrag;
    {
        BF16x16 t;
        #pragma unroll
        for (int i = 0; i < 16; ++i) t.u[i] = 0x3F80;
        onesfrag = t.v;
    }

    v8f acco[5];
    #pragma unroll
    for (int j = 0; j < 5; ++j) acco[j] = v8f{0,0,0,0,0,0,0,0};
    float mrow[8];
    #pragma unroll
    for (int r = 0; r < 8; ++r) mrow[r] = -3.0e38f;

    // per-lane fragment base pointers
    const unsigned short* krow = kbf + ((long)b * SEQ + lrow) * DIM + lhalf * 16; // + key*DIM + c*32
    const unsigned short* vrow = vtb + ((long)b * DIM + lrow) * SEQ + lhalf * 16; // + j*16*SEQ + kv0 + kc*32
    unsigned short* pw = &sP[wave * 16 * BN];

    for (int kt = 0; kt < SEQ / BN; ++kt) {
        const int kv0 = kt * BN;
        // prefetch next tile (global_prefetch_b8)
        if (kt + 1 < SEQ / BN) {
            __builtin_prefetch(krow + (long)(kv0 + BN + lane) * DIM, 0, 0);
            __builtin_prefetch(vrow + (long)(lane >> 2) * SEQ + kv0 + BN + (lane & 3) * 16, 0, 0);
        }

        // ---- scores: four 16x16 C tiles, K=64, B-fragments straight from global ----
        v8f s[4];
        #pragma unroll
        for (int h = 0; h < 4; ++h) {
            v8f acc = v8f{0,0,0,0,0,0,0,0};
            #pragma unroll
            for (int c = 0; c < 2; ++c) {
                v16bf bfrag = frag32B(krow + (long)(kv0 + h * 16) * DIM + c * 32);
                acc = __builtin_amdgcn_wmma_f32_16x16x32_bf16(
                        false, qfrag[c], false, bfrag, (short)0, acc, false, false);
            }
            s[h] = acc;
        }

        // ---- online softmax (log2 domain); only row-MAX reduces across lanes ----
        float mnew[8], alpha[8];
        #pragma unroll
        for (int r = 0; r < 8; ++r) {
            float c0 = fmaxf(fmaxf(s[0][r], s[1][r]), fmaxf(s[2][r], s[3][r]));
            c0 = fmaxf(c0, __shfl_xor(c0, 1, 16));
            c0 = fmaxf(c0, __shfl_xor(c0, 2, 16));
            c0 = fmaxf(c0, __shfl_xor(c0, 4, 16));
            c0 = fmaxf(c0, __shfl_xor(c0, 8, 16));
            mnew[r] = fmaxf(mrow[r], c0);
            alpha[r] = exp2f(mrow[r] - mnew[r]);
            mrow[r] = mnew[r];
        }
        #pragma unroll
        for (int h = 0; h < 4; ++h)
            #pragma unroll
            for (int r = 0; r < 8; ++r)
                pw[(r + 8*lhalf) * BN + h*16 + lrow] = f2bf(exp2f(s[h][r] - mnew[r]));
        #pragma unroll
        for (int j = 0; j < 5; ++j)
            #pragma unroll
            for (int r = 0; r < 8; ++r) acco[j][r] *= alpha[r];

        v16bf pfrag[2];
        #pragma unroll
        for (int kc = 0; kc < 2; ++kc)
            pfrag[kc] = frag2(&pw[lrow * BN + kc*32 + lhalf*16],
                              &pw[lrow * BN + kc*32 + lhalf*16 + 8]);

        // ---- O += P V (V^T fragments straight from global) ; lsum += P x ones ----
        #pragma unroll
        for (int kc = 0; kc < 2; ++kc) {
            #pragma unroll
            for (int j = 0; j < 4; ++j) {
                v16bf vfrag = frag32B(vrow + (long)(j * 16) * SEQ + kv0 + kc * 32);
                acco[j] = __builtin_amdgcn_wmma_f32_16x16x32_bf16(
                            false, pfrag[kc], false, vfrag, (short)0, acco[j], false, false);
            }
            acco[4] = __builtin_amdgcn_wmma_f32_16x16x32_bf16(
                        false, pfrag[kc], false, onesfrag, (short)0, acco[4], false, false);
        }
    }

    float inv[8];
    #pragma unroll
    for (int r = 0; r < 8; ++r) inv[r] = 1.0f / acco[4][r];
    float* op = out + ((long)b * SEQ + qbase + wave * 16) * DIM;
    #pragma unroll
    for (int j = 0; j < 4; ++j)
        #pragma unroll
        for (int r = 0; r < 8; ++r)
            op[(r + 8*lhalf) * DIM + j*16 + lrow] = acco[j][r] * inv[r];
}

// ============== fallback (self-contained, converts in-loop) ==============
__global__ __launch_bounds__(128)
void fa_fwd_fallback(const float* __restrict__ q, const float* __restrict__ k,
                     const float* __restrict__ v, const float* __restrict__ inv_scale,
                     float* __restrict__ out)
{
    __shared__ __align__(16) unsigned short sK [BN * DIM];
    __shared__ __align__(16) unsigned short sVT[DIM * BN];
    __shared__ __align__(16) unsigned short sP [NW * 16 * BN];

    const int b     = blockIdx.y;
    const int qbase = blockIdx.x * BM;
    const int tid   = threadIdx.x;
    const int wave  = tid >> 5;
    const int lane  = tid & 31;
    const int lrow  = lane & 15;
    const int lhalf = lane >> 4;

    const float qscale = (1.0f / inv_scale[0]) * 1.44269504088896340736f;

    const float* qp = q + ((long)b * SEQ + qbase + wave * 16 + lrow) * DIM;
    v16bf qfrag[2];
    for (int c = 0; c < 2; ++c) {
        const float4* p0 = (const float4*)(qp + c * 32 + lhalf * 16);
        const float4* p1 = (const float4*)(qp + c * 32 + lhalf * 16 + 8);
        float4 a0 = p0[0], a1 = p0[1], b0 = p1[0], b1 = p1[1];
        float f0[8] = {a0.x,a0.y,a0.z,a0.w,a1.x,a1.y,a1.z,a1.w};
        float f1[8] = {b0.x,b0.y,b0.z,b0.w,b1.x,b1.y,b1.z,b1.w};
        BF16x16 t;
        #pragma unroll
        for (int i = 0; i < 8; ++i) {
            t.u[i]     = f2bf(f0[i] * qscale);
            t.u[8 + i] = f2bf(f1[i] * qscale);
        }
        qfrag[c] = t.v;
    }
    v16bf onesfrag;
    {
        BF16x16 t;
        #pragma unroll
        for (int i = 0; i < 16; ++i) t.u[i] = 0x3F80;
        onesfrag = t.v;
    }
    v8f acco[5];
    #pragma unroll
    for (int j = 0; j < 5; ++j) acco[j] = v8f{0,0,0,0,0,0,0,0};
    float mrow[8];
    #pragma unroll
    for (int r = 0; r < 8; ++r) mrow[r] = -3.0e38f;

    const int  key    = tid >> 1;
    const int  dpart  = (tid & 1) * 32;
    const long kvbase = (long)b * SEQ * DIM;
    unsigned short* pw = &sP[wave * 16 * BN];

    for (int kt = 0; kt < SEQ / BN; ++kt) {
        const int kv0 = kt * BN;
        {
            const float4* k4 = (const float4*)(k + kvbase + (long)(kv0 + key) * DIM + dpart);
            BF16x16 t[2];
            #pragma unroll
            for (int h = 0; h < 2; ++h) {
                float f[16];
                #pragma unroll
                for (int i = 0; i < 4; ++i) {
                    float4 x = k4[4*h + i];
                    f[4*i] = x.x; f[4*i+1] = x.y; f[4*i+2] = x.z; f[4*i+3] = x.w;
                }
                #pragma unroll
                for (int i = 0; i < 16; ++i) t[h].u[i] = f2bf(f[i]);
                __builtin_memcpy(&sK[key * DIM + dpart + 16*h], t[h].q, 32);
            }
        }
        {
            const float4* v4 = (const float4*)(v + kvbase + (long)(kv0 + key) * DIM + dpart);
            #pragma unroll
            for (int h = 0; h < 8; ++h) {
                float4 x = v4[h];
                sVT[(dpart + 4*h + 0) * BN + key] = f2bf(x.x);
                sVT[(dpart + 4*h + 1) * BN + key] = f2bf(x.y);
                sVT[(dpart + 4*h + 2) * BN + key] = f2bf(x.z);
                sVT[(dpart + 4*h + 3) * BN + key] = f2bf(x.w);
            }
        }
        __syncthreads();
        v8f s[4];
        #pragma unroll
        for (int h = 0; h < 4; ++h) {
            v8f acc = v8f{0,0,0,0,0,0,0,0};
            #pragma unroll
            for (int c = 0; c < 2; ++c) {
                v16bf bfrag = frag2(&sK[(h*16 + lrow) * DIM + c*32 + lhalf*16],
                                    &sK[(h*16 + lrow) * DIM + c*32 + lhalf*16 + 8]);
                acc = __builtin_amdgcn_wmma_f32_16x16x32_bf16(
                        false, qfrag[c], false, bfrag, (short)0, acc, false, false);
            }
            s[h] = acc;
        }
        float mnew[8], alpha[8];
        #pragma unroll
        for (int r = 0; r < 8; ++r) {
            float c0 = fmaxf(fmaxf(s[0][r], s[1][r]), fmaxf(s[2][r], s[3][r]));
            c0 = fmaxf(c0, __shfl_xor(c0, 1, 16));
            c0 = fmaxf(c0, __shfl_xor(c0, 2, 16));
            c0 = fmaxf(c0, __shfl_xor(c0, 4, 16));
            c0 = fmaxf(c0, __shfl_xor(c0, 8, 16));
            mnew[r] = fmaxf(mrow[r], c0);
            alpha[r] = exp2f(mrow[r] - mnew[r]);
            mrow[r] = mnew[r];
        }
        #pragma unroll
        for (int h = 0; h < 4; ++h)
            #pragma unroll
            for (int r = 0; r < 8; ++r)
                pw[(r + 8*lhalf) * BN + h*16 + lrow] = f2bf(exp2f(s[h][r] - mnew[r]));
        #pragma unroll
        for (int j = 0; j < 5; ++j)
            #pragma unroll
            for (int r = 0; r < 8; ++r) acco[j][r] *= alpha[r];
        v16bf pfrag[2];
        #pragma unroll
        for (int kc = 0; kc < 2; ++kc)
            pfrag[kc] = frag2(&pw[lrow * BN + kc*32 + lhalf*16],
                              &pw[lrow * BN + kc*32 + lhalf*16 + 8]);
        #pragma unroll
        for (int kc = 0; kc < 2; ++kc) {
            #pragma unroll
            for (int j = 0; j < 4; ++j) {
                v16bf vfrag = frag2(&sVT[(j*16 + lrow) * BN + kc*32 + lhalf*16],
                                    &sVT[(j*16 + lrow) * BN + kc*32 + lhalf*16 + 8]);
                acco[j] = __builtin_amdgcn_wmma_f32_16x16x32_bf16(
                            false, pfrag[kc], false, vfrag, (short)0, acco[j], false, false);
            }
            acco[4] = __builtin_amdgcn_wmma_f32_16x16x32_bf16(
                        false, pfrag[kc], false, onesfrag, (short)0, acco[4], false, false);
        }
        __syncthreads();
    }
    float inv[8];
    #pragma unroll
    for (int r = 0; r < 8; ++r) inv[r] = 1.0f / acco[4][r];
    float* op = out + ((long)b * SEQ + qbase + wave * 16) * DIM;
    #pragma unroll
    for (int j = 0; j < 4; ++j)
        #pragma unroll
        for (int r = 0; r < 8; ++r)
            op[(r + 8*lhalf) * DIM + j*16 + lrow] = acco[j][r] * inv[r];
}

extern "C" void kernel_launch(void* const* d_in, const int* in_sizes, int n_in,
                              void* d_out, int out_size, void* d_ws, size_t ws_size,
                              hipStream_t stream) {
    const float* q   = (const float*)d_in[0];
    const float* k   = (const float*)d_in[1];
    const float* v   = (const float*)d_in[2];
    const float* inv = (const float*)d_in[3];
    (void)in_sizes; (void)n_in; (void)out_size;
    float* out = (float*)d_out;

    const size_t elems  = (size_t)BATCH * SEQ * DIM;
    const size_t needed = 2 * elems * sizeof(unsigned short); // K bf16 + V^T bf16

    if (d_ws != nullptr && ws_size >= needed) {
        unsigned short* kbf = (unsigned short*)d_ws;
        unsigned short* vtb = kbf + elems;
        conv_k_bf16<<<dim3((unsigned)(elems / (256 * 8))), dim3(256), 0, stream>>>(k, kbf);
        conv_v_t_bf16<<<dim3(SEQ / 64, BATCH), dim3(256), 0, stream>>>(v, vtb);
        fa_fwd_fast<<<dim3(SEQ / BM, BATCH), dim3(128), 0, stream>>>(q, kbf, vtb, inv, out);
    } else {
        fa_fwd_fallback<<<dim3(SEQ / BM, BATCH), dim3(128), 0, stream>>>(q, k, v, inv, out);
    }
}
